// LabelGuidedPatchSelector_68092411510808
// MI455X (gfx1250) — matrix operand order repaired and checked by reading, not compiled, not run on hardware
//
#include <hip/hip_runtime.h>
#include <hip/hip_bf16.h>
#include <math.h>

typedef __attribute__((ext_vector_type(16))) __bf16 v16bf;
typedef __attribute__((ext_vector_type(8)))  float  v8f;

#define B_DIM   32
#define N_PATCH 4096
#define D_VIS   768
#define HID     256
#define KTILES  24          // 768 / 32
#define NTILES  16          // 256 / 16
#define KCHUNK  (32 * HID)  // bf16 elements per staged k-chunk (16 KB)
#define TOPK    192
#define VIS_W   0.4f
#define LBL_W   0.6f

// -------------------------------------------------------------------------
// Prep 1: W1 (fp32 [768,256]) -> bf16 fragment-contiguous layout
// w1f[((kt*16 + nt)*32 + lane)*16 + j] = W1[(kt*32 + kk)*256 + (nt*16 + lane%16)]
//   B-frag (32x16 bf16) layout: lane = column, lanes 0-15 hold K=0-7/16-23,
//   lanes 16-31 hold K=8-15/24-31 (matches A-frag order).
// -------------------------------------------------------------------------
__global__ __launch_bounds__(256)
void prep_w1_kernel(const float* __restrict__ W1, __bf16* __restrict__ w1f) {
    int idx = blockIdx.x * 256 + threadIdx.x;   // 24*16*32*16 = 196608 total
    int j   = idx & 15;
    int ln  = (idx >> 4) & 31;
    int nt  = (idx >> 9) & 15;
    int kt  = idx >> 13;
    int kk  = ((ln < 16) ? 0 : 8) + ((j < 8) ? j : j + 8);
    int col = nt * 16 + (ln & 15);
    int kg  = kt * 32 + kk;
    w1f[idx] = (__bf16)W1[kg * HID + col];
}

// -------------------------------------------------------------------------
// Prep 2: label_proj = label_context @ Wp + bp, then L2-normalize -> u[B,768]
// -------------------------------------------------------------------------
__global__ __launch_bounds__(256)
void prep_label_kernel(const float* __restrict__ lbl, const float* __restrict__ Wp,
                       const float* __restrict__ bp, float* __restrict__ u) {
    __shared__ float proj[D_VIS];
    __shared__ float red[256];
    const int b = blockIdx.x;
    const int t = threadIdx.x;
    float ss = 0.f;
    for (int i = t; i < D_VIS; i += 256) {
        float acc = bp[i];
        for (int kk = 0; kk < 128; ++kk)
            acc = fmaf(lbl[b * 128 + kk], Wp[kk * D_VIS + i], acc);
        proj[i] = acc;
        ss += acc * acc;
    }
    red[t] = ss;
    __syncthreads();
    for (int s = 128; s > 0; s >>= 1) {
        if (t < s) red[t] += red[t + s];
        __syncthreads();
    }
    const float inv = 1.0f / fmaxf(sqrtf(red[0]), 1e-12f);
    for (int i = t; i < D_VIS; i += 256) u[b * D_VIS + i] = proj[i] * inv;
}

// -------------------------------------------------------------------------
// Main fused scoring kernel: 8 waves/block, one 16-row M-tile per wave.
// W1 k-chunks double-buffered into LDS via global_load_async_to_lds_b128.
// -------------------------------------------------------------------------
__global__ __launch_bounds__(256)
void score_kernel(const float* __restrict__ feats, const __bf16* __restrict__ w1f,
                  const float* __restrict__ b1, const float* __restrict__ W2,
                  const float* __restrict__ b2, const float* __restrict__ u,
                  const unsigned char* __restrict__ mask, float* __restrict__ scores) {
    __shared__ __attribute__((aligned(32))) __bf16 lbuf[2][KCHUNK];   // 32 KB

    const int tid  = threadIdx.x;
    const int lane = tid & 31;
    const int wave = tid >> 5;
    const int b    = blockIdx.x >> 5;
    const int mg   = blockIdx.x & 31;
    const int m0   = (mg * 8 + wave) * 16;
    const int r    = lane & 15;
    const int hi   = lane >> 4;

    const float* frow = feats + ((size_t)(b * N_PATCH + m0 + r)) * D_VIS;
    const float* urow = u + (size_t)b * D_VIS;

    v8f acc[NTILES] = {};
    float dotAcc = 0.f, nrmAcc = 0.f;

    // stage chunk 0 into lbuf[0]: 16 KB / 256 threads = 64 B/thread
    {
        const char* src   = (const char*)w1f;
        unsigned    lbase = (unsigned)(unsigned long long)(&lbuf[0][0]);
#pragma unroll
        for (int i = 0; i < 4; ++i) {
            unsigned loff = lbase + tid * 64u + i * 16u;
            unsigned goff = tid * 64u + i * 16u;
            asm volatile("global_load_async_to_lds_b128 %0, %1, %2"
                         :: "v"(loff), "v"(goff), "s"(src) : "memory");
        }
    }

    for (int kt = 0; kt < KTILES; ++kt) {
        if (kt + 1 < KTILES) {
            const char* src   = (const char*)(w1f + (size_t)(kt + 1) * KCHUNK);
            unsigned    lbase = (unsigned)(unsigned long long)(&lbuf[(kt + 1) & 1][0]);
#pragma unroll
            for (int i = 0; i < 4; ++i) {
                unsigned loff = lbase + tid * 64u + i * 16u;
                unsigned goff = tid * 64u + i * 16u;
                asm volatile("global_load_async_to_lds_b128 %0, %1, %2"
                             :: "v"(loff), "v"(goff), "s"(src) : "memory");
            }
            // 4 newer ops outstanding; chunk kt's 4 are the older in-order ops
            asm volatile("s_wait_asynccnt 4" ::: "memory");
            __builtin_prefetch(frow + (kt + 1) * 32 + hi * 8, 0, 1);
        } else {
            asm volatile("s_wait_asynccnt 0" ::: "memory");
        }
        __syncthreads();   // chunk kt resident for all waves

        // ---- A fragment (ISA 16-bit A 16x32 layout) + label-score partials
        const int kbase = kt * 32 + hi * 8;   // lanes<16: K 0-7/16-23 ; lanes>=16: 8-15/24-31
        float4 f0 = *(const float4*)(frow + kbase + 0);
        float4 f1 = *(const float4*)(frow + kbase + 4);
        float4 f2 = *(const float4*)(frow + kbase + 16);
        float4 f3 = *(const float4*)(frow + kbase + 20);
        float4 u0 = *(const float4*)(urow + kbase + 0);
        float4 u1 = *(const float4*)(urow + kbase + 4);
        float4 u2 = *(const float4*)(urow + kbase + 16);
        float4 u3 = *(const float4*)(urow + kbase + 20);

        float xv[16] = {f0.x, f0.y, f0.z, f0.w, f1.x, f1.y, f1.z, f1.w,
                        f2.x, f2.y, f2.z, f2.w, f3.x, f3.y, f3.z, f3.w};
        float uv[16] = {u0.x, u0.y, u0.z, u0.w, u1.x, u1.y, u1.z, u1.w,
                        u2.x, u2.y, u2.z, u2.w, u3.x, u3.y, u3.z, u3.w};
        v16bf afrag;
#pragma unroll
        for (int j = 0; j < 16; ++j) {
            afrag[j] = (__bf16)xv[j];
            dotAcc = fmaf(xv[j], uv[j], dotAcc);
            nrmAcc = fmaf(xv[j], xv[j], nrmAcc);
        }

        // ---- 16 hidden tiles: single 32-B LDS read per lane per B-frag
        const __bf16* lchunk = &lbuf[kt & 1][0];
#pragma unroll
        for (int nt = 0; nt < NTILES; ++nt) {
            v16bf bfrag = *(const v16bf*)(lchunk + (nt * 32 + lane) * 16);
            acc[nt] = __builtin_amdgcn_wmma_f32_16x16x32_bf16(
                false, afrag, false, bfrag, (short)0, acc[nt], false, false);
        }
        __syncthreads();   // all waves done with lbuf[kt&1] before restaging
    }

    // ---- epilogue: relu(h + b1) . W2, reduce over hidden (lanes within half)
    float vs[8];
#pragma unroll
    for (int j = 0; j < 8; ++j) vs[j] = 0.f;
#pragma unroll
    for (int nt = 0; nt < NTILES; ++nt) {
        const int n   = nt * 16 + r;
        const float bias = b1[n];
        const float w2v  = W2[n];
#pragma unroll
        for (int j = 0; j < 8; ++j) {
            float v = acc[nt][j] + bias;
            v = v > 0.f ? v : 0.f;
            vs[j] = fmaf(v, w2v, vs[j]);
        }
    }
#pragma unroll
    for (int off = 1; off < 16; off <<= 1) {
#pragma unroll
        for (int j = 0; j < 8; ++j) vs[j] += __shfl_xor(vs[j], off, 32);
    }
    // label score: combine the two half-lane K-partials of each row
    float dt  = dotAcc + __shfl_xor(dotAcc, 16, 32);
    float nm  = nrmAcc + __shfl_xor(nrmAcc, 16, 32);
    float lab = dt / fmaxf(sqrtf(nm), 1e-12f);   // lane L holds row (L & 15)

    const int jrow = lane & 7;
    float myvis = vs[0];
#pragma unroll
    for (int j = 1; j < 8; ++j)
        if (jrow == j) myvis = vs[j];
    const int   row_in = hi * 8 + jrow;          // C layout: half0 rows 0-7, half1 rows 8-15
    const float mylab  = __shfl(lab, row_in, 32);
    if ((lane & 15) < 8) {
        const int row = m0 + row_in;
        float c = VIS_W * (myvis + b2[0]) + LBL_W * mylab;
        if (!mask[(size_t)b * N_PATCH + row]) c = -INFINITY;
        scores[(size_t)b * N_PATCH + row] = c;
    }
}

// -------------------------------------------------------------------------
// Top-k: per-batch bitonic sort of 4096 packed (score,index) keys in LDS.
// Key = monotone-uint(score) << 12 | (4095 - idx)  -> descending sort gives
// jax.lax.top_k order with lower-index-first ties.
// -------------------------------------------------------------------------
__global__ __launch_bounds__(256)
void topk_kernel(const float* __restrict__ scores, int* __restrict__ out_idx) {
    __shared__ unsigned long long keys[N_PATCH];   // 32 KB
    const int b = blockIdx.x;
    const int t = threadIdx.x;
    for (int i = t; i < N_PATCH; i += 256) {
        unsigned uu = __float_as_uint(scores[(size_t)b * N_PATCH + i]);
        uu = (uu & 0x80000000u) ? ~uu : (uu | 0x80000000u);
        keys[i] = ((unsigned long long)uu << 12) | (unsigned)(N_PATCH - 1 - i);
    }
    __syncthreads();
    for (int kk = 2; kk <= N_PATCH; kk <<= 1) {
        for (int jj = kk >> 1; jj > 0; jj >>= 1) {
            for (int i = t; i < N_PATCH; i += 256) {
                const int ixj = i ^ jj;
                if (ixj > i) {
                    const bool up = (i & kk) == 0;    // descending overall
                    unsigned long long a = keys[i], c = keys[ixj];
                    if (up ? (a < c) : (a > c)) { keys[i] = c; keys[ixj] = a; }
                }
            }
            __syncthreads();
        }
    }
    for (int i = t; i < TOPK; i += 256)
        out_idx[b * TOPK + i] = N_PATCH - 1 - (int)(keys[i] & 0xFFFu);
}

// -------------------------------------------------------------------------
// Gather: selected_feats[b, s, :] = visual_feats[b, idx[b,s], :]
// -------------------------------------------------------------------------
__global__ __launch_bounds__(256)
void gather_kernel(const float* __restrict__ feats, const int* __restrict__ idx,
                   float* __restrict__ out) {
    const int b = blockIdx.x / TOPK;
    const int s = blockIdx.x % TOPK;
    const int src = idx[b * TOPK + s];
    const float* sp = feats + ((size_t)b * N_PATCH + src) * D_VIS;
    float* dp = out + ((size_t)b * TOPK + s) * D_VIS;
    for (int i = threadIdx.x; i < D_VIS; i += 256) dp[i] = sp[i];
}

// -------------------------------------------------------------------------
extern "C" void kernel_launch(void* const* d_in, const int* in_sizes, int n_in,
                              void* d_out, int out_size, void* d_ws, size_t ws_size,
                              hipStream_t stream) {
    const float*         feats = (const float*)d_in[0];
    const float*         lbl   = (const float*)d_in[1];
    const float*         W1    = (const float*)d_in[2];
    const float*         b1    = (const float*)d_in[3];
    const float*         W2    = (const float*)d_in[4];
    const float*         b2    = (const float*)d_in[5];
    const float*         Wp    = (const float*)d_in[6];
    const float*         bp    = (const float*)d_in[7];
    const unsigned char* mask  = (const unsigned char*)d_in[8];
    // d_in[9] = k (on device); k=192 is fixed by out_size.

    float* out_feats = (float*)d_out;
    int*   out_idx   = (int*)(out_feats + (size_t)B_DIM * TOPK * D_VIS);

    char*   ws     = (char*)d_ws;
    float*  u      = (float*)ws;                                    //  96 KB
    __bf16* w1f    = (__bf16*)(ws + (size_t)B_DIM * D_VIS * 4);     // 384 KB
    float*  scores = (float*)(ws + (size_t)B_DIM * D_VIS * 4
                                 + (size_t)D_VIS * HID * 2);        // 512 KB

    prep_w1_kernel   <<<(KTILES * NTILES * 32 * 16) / 256, 256, 0, stream>>>(W1, w1f);
    prep_label_kernel<<<B_DIM, 256, 0, stream>>>(lbl, Wp, bp, u);
    score_kernel     <<<B_DIM * (N_PATCH / (16 * 8)), 256, 0, stream>>>(
        feats, w1f, b1, W2, b2, u, mask, scores);
    topk_kernel      <<<B_DIM, 256, 0, stream>>>(scores, out_idx);
    gather_kernel    <<<B_DIM * TOPK, 256, 0, stream>>>(feats, out_idx, out_feats);
}
